// GCNRegression_21165598834730
// MI455X (gfx1250) — matrix operand order repaired and checked by reading, not compiled, not run on hardware
//
#include <hip/hip_runtime.h>

#define D 128
#define APAD 136     // LDS row pitch (bf16 elems): 272B -> +4 banks/row, conflict-free
#define MTILE 32     // rows per block-iteration (2 WMMA sub-tiles in M)

typedef __attribute__((ext_vector_type(16))) __bf16 v16bf;
typedef __attribute__((ext_vector_type(8)))  float  v8f;

union V16 { v16bf v; uint4 q[2]; };

__device__ __forceinline__ unsigned short f2bf(float f) {
  unsigned u = __float_as_uint(f);
  u += 0x7FFFu + ((u >> 16) & 1u);     // round-to-nearest-even
  return (unsigned short)(u >> 16);
}

// ---- graph-structure precompute -------------------------------------------

__global__ void k_cvt_w(const float* __restrict__ w, unsigned short* __restrict__ o, int n) {
  int i = blockIdx.x * 256 + threadIdx.x;
  if (i < n) o[i] = f2bf(w[i]);
}

__global__ void k_fill1(float* __restrict__ d, int n) {
  int i = blockIdx.x * 256 + threadIdx.x;
  if (i < n) d[i] = 1.0f;                       // self-loop contributes 1 to degree
}

__global__ void k_deg(const int* __restrict__ dst, float* __restrict__ deg, int e) {
  int i = blockIdx.x * 256 + threadIdx.x;
  if (i < e) atomicAdd(&deg[dst[i]], 1.0f);
}

__global__ void k_dinv(const float* __restrict__ deg, float* __restrict__ dinv, int n) {
  int i = blockIdx.x * 256 + threadIdx.x;
  if (i < n) dinv[i] = rsqrtf(deg[i]);          // deg >= 1 always (self loops)
}

__global__ void k_norm(const int* __restrict__ src, const int* __restrict__ dst,
                       const float* __restrict__ dinv, float* __restrict__ nrm, int e) {
  int i = blockIdx.x * 256 + threadIdx.x;
  if (i < e) nrm[i] = dinv[src[i]] * dinv[dst[i]];
}

// ---- persistent WMMA GEMM: Hw = H @ W ; Agg = Hw * dinv(row)^2 ------------
// Safe when Agg aliases H: the grid-stride tile loop partitions rows, and a
// block stages its rows into LDS (barrier) before writing them back.
// B fragments (W columns) are kernel-invariant: loaded to registers ONCE.

__global__ void k_gemm(const float* __restrict__ H, const unsigned short* __restrict__ Wb,
                       float* __restrict__ Hw, float* __restrict__ Agg,
                       const float* __restrict__ dinv, int n, int numTiles) {
  __shared__ unsigned short lW[D * APAD];       // W transposed: lW[col][k], staged once
  __shared__ unsigned short lA[MTILE * APAD];   // A tile, row-major

  const int t = threadIdx.x;                    // 256 threads = 8 waves

  // stage W transposed once per block (packed uint loads: 2 bf16 per load)
  {
    const unsigned* Wb2 = (const unsigned*)Wb;
    for (int idx = t; idx < D * D / 2; idx += 256) {
      unsigned u = Wb2[idx];
      int k = idx >> 6;                         // 64 uints per W row
      int c = (idx & 63) * 2;
      lW[c * APAD + k]       = (unsigned short)u;
      lW[(c + 1) * APAD + k] = (unsigned short)(u >> 16);
    }
  }
  __syncthreads();

  const int wave = t >> 5, lane = t & 31;
  const int hh = lane >> 4, lr = lane & 15;     // hh: 16-lane half
  const int n0 = wave * 16;                     // wave owns a 16-col slab

  // hoist B fragments into registers for the whole kernel
  // 16-bit B layout: lanes 0-15: K=kb+0..7 / kb+16..23; lanes 16-31: +8
  V16 bfr[4];
  {
    const unsigned short* bBase = lW + (n0 + lr) * APAD;
#pragma unroll
    for (int kk = 0; kk < 4; ++kk) {
      const int kb = kk * 32;
      bfr[kk].q[0] = *(const uint4*)(bBase + kb + hh * 8);
      bfr[kk].q[1] = *(const uint4*)(bBase + kb + 16 + hh * 8);
    }
  }

  for (int tile = blockIdx.x; tile < numTiles; tile += gridDim.x) {
    const int m0 = tile * MTILE;
    __syncthreads();                            // lA readers from prev iter done

    // stage + convert A tile: float4 global loads, packed ds_store_b64 writes
    for (int idx = t; idx < MTILE * D / 4; idx += 256) {
      int r = idx >> 5;                         // 32 float4 per row
      int c = (idx & 31) * 4;
      int row = m0 + r;
      float4 v = make_float4(0.f, 0.f, 0.f, 0.f);
      if (row < n) v = ((const float4*)H)[(size_t)row * (D / 4) + (idx & 31)];
      unsigned long long pk =  (unsigned long long)f2bf(v.x)
                            | ((unsigned long long)f2bf(v.y) << 16)
                            | ((unsigned long long)f2bf(v.z) << 32)
                            | ((unsigned long long)f2bf(v.w) << 48);
      *(unsigned long long*)(lA + r * APAD + c) = pk;
    }

    // prefetch next grid-stride tile's rows into cache (global_prefetch_b8)
    {
      int nt = tile + gridDim.x;
      if (nt < numTiles && t < MTILE) {
        int nr = nt * MTILE + t;
        if (nr < n) __builtin_prefetch(H + (size_t)nr * D, 0, 3);
      }
    }
    __syncthreads();

    const bool full = (m0 + MTILE) <= n;        // block-uniform fast path
    const int col = n0 + lr;
#pragma unroll
    for (int mt = 0; mt < 2; ++mt) {            // two 16-row sub-tiles
      const unsigned short* aBase = lA + (mt * 16 + lr) * APAD;
      v8f acc = {};
#pragma unroll
      for (int kk = 0; kk < 4; ++kk) {          // K = 128 in 4 steps of 32
        const int kb = kk * 32;
        V16 a;
        // 16-bit A layout: lanes 0-15: K=kb+0..7 / kb+16..23; lanes 16-31: +8
        a.q[0] = *(const uint4*)(aBase + kb + hh * 8);
        a.q[1] = *(const uint4*)(aBase + kb + 16 + hh * 8);
        acc = __builtin_amdgcn_wmma_f32_16x16x32_bf16(false, a.v, false, bfr[kk].v,
                                                      (short)0, acc, false, false);
      }
      // C/D layout: VGPR r -> row rbase + r, lane -> col n0 + lr
      const int rbase = m0 + mt * 16 + hh * 8;
      if (full) {
        float*       hp = Hw   + (size_t)rbase * D + col;   // const 512B row stride ->
        float*       ap = Agg  + (size_t)rbase * D + col;   // immediate-offset stores
        const float* dp = dinv + rbase;
#pragma unroll
        for (int r = 0; r < 8; ++r) {
          float v = acc[r], di = dp[r];
          hp[r * D] = v;
          ap[r * D] = v * di * di;              // fused self-loop init
        }
      } else {
        for (int r = 0; r < 8; ++r) {
          int row = rbase + r;
          if (row < n) {
            float v = acc[r], di = dinv[row];
            Hw[(size_t)row * D + col]  = v;
            Agg[(size_t)row * D + col] = v * di * di;
          }
        }
      }
    }
  }
}

// ---- aggregation ----------------------------------------------------------

__global__ void k_scatter2(const float2* __restrict__ hw2, const int* __restrict__ src,
                           const int* __restrict__ dst, const float* __restrict__ nrm,
                           float* __restrict__ agg, long long tot2) {
  long long i = (long long)blockIdx.x * 256 + threadIdx.x;
  if (i >= tot2) return;
  int e = (int)(i >> 6), d2 = (int)(i & 63);    // 64 consecutive lanes = one edge row
  float nv = nrm[e];
  float2 v = hw2[(size_t)src[e] * (D / 2) + d2];
  float* ap = agg + (size_t)dst[e] * D + d2 * 2;
  atomicAdd(ap,     v.x * nv);
  atomicAdd(ap + 1, v.y * nv);
}

__global__ void k_biasrelu4(float4* __restrict__ h, const float* __restrict__ b, long long tot4) {
  long long i = (long long)blockIdx.x * 256 + threadIdx.x;
  if (i >= tot4) return;
  float4 v = h[i];
  const float* bp = b + (int)(i & 31) * 4;
  v.x = fmaxf(v.x + bp[0], 0.f);
  v.y = fmaxf(v.y + bp[1], 0.f);
  v.z = fmaxf(v.z + bp[2], 0.f);
  v.w = fmaxf(v.w + bp[3], 0.f);
  h[i] = v;
}

// ---- pooling + readout ----------------------------------------------------

__global__ void k_zero4(float4* __restrict__ d, int n4) {
  int i = blockIdx.x * 256 + threadIdx.x;
  if (i < n4) d[i] = make_float4(0.f, 0.f, 0.f, 0.f);
}

__global__ void k_count(const int* __restrict__ batch, float* __restrict__ counts, int n) {
  int i = blockIdx.x * 256 + threadIdx.x;
  if (i < n) atomicAdd(&counts[batch[i]], 1.0f);
}

__global__ void k_pool2(const float2* __restrict__ h2, const int* __restrict__ batch,
                        float* __restrict__ pooled, long long tot2) {
  long long i = (long long)blockIdx.x * 256 + threadIdx.x;
  if (i >= tot2) return;
  int node = (int)(i >> 6), d2 = (int)(i & 63);
  float2 v = h2[i];
  float* pp = pooled + (size_t)batch[node] * D + d2 * 2;
  atomicAdd(pp,     v.x);
  atomicAdd(pp + 1, v.y);
}

__global__ void k_final(const float* __restrict__ pooled, const float* __restrict__ counts,
                        const float* __restrict__ lw, const float* __restrict__ lb,
                        float* __restrict__ out) {
  __shared__ float red[D];
  int g = blockIdx.x, t = threadIdx.x;
  float c = fmaxf(counts[g], 1.0f);
  red[t] = (pooled[g * D + t] / c) * lw[t];
  __syncthreads();
  for (int s = 64; s > 0; s >>= 1) {
    if (t < s) red[t] += red[t + s];
    __syncthreads();
  }
  if (t == 0) out[g] = red[0] + lb[0];
}

// ---------------------------------------------------------------------------

extern "C" void kernel_launch(void* const* d_in, const int* in_sizes, int n_in,
                              void* d_out, int out_size, void* d_ws, size_t ws_size,
                              hipStream_t stream) {
  const float* x      = (const float*)d_in[0];
  const int*   eidx   = (const int*)  d_in[1];
  const int*   batch  = (const int*)  d_in[2];
  const float* Ws     = (const float*)d_in[3];
  const float* bs     = (const float*)d_in[4];
  const float* lin_w  = (const float*)d_in[5];
  const float* lin_b  = (const float*)d_in[6];
  float*       out    = (float*)d_out;

  const int Nn = in_sizes[0] / D;
  const int E  = in_sizes[1] / 2;
  const int L  = in_sizes[3] / (D * D);
  const int G  = out_size;
  const int* src = eidx;
  const int* dst = eidx + E;

  // carve workspace (256B aligned)
  char* p = (char*)d_ws;
  auto carve = [&](size_t bytes) -> char* {
    char* r = p; p += (bytes + 255) & ~(size_t)255; return r;
  };
  float*          deg    = (float*)carve((size_t)Nn * 4);
  float*          dinv   = (float*)carve((size_t)Nn * 4);
  float*          normE  = (float*)carve((size_t)E * 4);
  float*          buf0   = (float*)carve((size_t)Nn * D * 4);   // agg / h
  float*          buf1   = (float*)carve((size_t)Nn * D * 4);   // hw
  unsigned short* Wbf    = (unsigned short*)carve((size_t)L * D * D * 2);
  float*          pooled = (float*)carve((size_t)G * D * 4);
  float*          counts = (float*)carve((size_t)G * 4);
  (void)ws_size; (void)n_in;

  const long long totND = (long long)Nn * D;
  const long long totED = (long long)E * D;
  const int numTiles = (Nn + MTILE - 1) / MTILE;
  const unsigned gemmBlocks = (unsigned)(numTiles < 512 ? numTiles : 512);
  auto blk = [](long long tot) { return (unsigned)((tot + 255) / 256); };

  // precompute
  k_cvt_w<<<blk(L * D * D), 256, 0, stream>>>(Ws, Wbf, L * D * D);
  k_fill1<<<blk(Nn), 256, 0, stream>>>(deg, Nn);
  k_deg  <<<blk(E),  256, 0, stream>>>(dst, deg, E);
  k_dinv <<<blk(Nn), 256, 0, stream>>>(deg, dinv, Nn);
  k_norm <<<blk(E),  256, 0, stream>>>(src, dst, dinv, normE, E);

  // conv layers: {hw, agg=hw*dinv^2} = gemm(cur) -> scatter-add -> bias+relu
  for (int l = 0; l < L; ++l) {
    const float* cur = (l == 0) ? x : buf0;
    k_gemm     <<<gemmBlocks, 256, 0, stream>>>(cur, Wbf + (size_t)l * D * D,
                                                buf1, buf0, dinv, Nn, numTiles);
    k_scatter2 <<<blk(totED / 2), 256, 0, stream>>>((const float2*)buf1, src, dst,
                                                    normE, buf0, totED / 2);
    k_biasrelu4<<<blk(totND / 4), 256, 0, stream>>>((float4*)buf0,
                                                    bs + (size_t)l * D, totND / 4);
  }

  // global mean pool + linear readout
  k_zero4<<<blk(G * D / 4), 256, 0, stream>>>((float4*)pooled, G * D / 4);
  k_zero4<<<blk((G + 3) / 4), 256, 0, stream>>>((float4*)counts, (G + 3) / 4);
  k_count<<<blk(Nn), 256, 0, stream>>>(batch, counts, Nn);
  k_pool2<<<blk(totND / 2), 256, 0, stream>>>((const float2*)buf0, batch, pooled, totND / 2);
  k_final<<<(unsigned)G, D, 0, stream>>>(pooled, counts, lin_w, lin_b, out);
}
